// seq2seq_fuelcell_21827023798500
// MI455X (gfx1250) — matrix operand order repaired and checked by reading, not compiled, not run on hardware
//
#include <hip/hip_runtime.h>
#include <hip/hip_bf16.h>
#include <stdint.h>

typedef _Float16 f16;
typedef __attribute__((ext_vector_type(16))) _Float16 v16h;
typedef __attribute__((ext_vector_type(8)))  _Float16 v8h;
typedef __attribute__((ext_vector_type(8)))  float    v8f;
typedef __attribute__((ext_vector_type(4)))  int      v4i;

#define B_TOT   4096
#define S_LEN   128
#define ENCH    128
#define DECH    256
#define ELDK    160   // encoder augmented K: 128 h | 5 x | 3 zero | 1 bias | pad
#define BIASK   136   // constant-1 column (after 16B-aligned x slot 128..135)
#define CLDK    160   // context augmented K: 128 ctx | 1 | pad
#define DLDK    288   // decoder augmented K: 256 h | 1 | pad
#define MT      64    // encoder batch rows per workgroup

#define AS1 __attribute__((address_space(1)))
#define AS3 __attribute__((address_space(3)))

#if __has_builtin(__builtin_amdgcn_global_load_async_to_lds_b128) && \
    __has_builtin(__builtin_amdgcn_s_wait_asynccnt)
#define USE_ASYNC_LDS 1
#endif

// ---------------------------------------------------------------- WMMA utils
__device__ __forceinline__ v8f wmma16(v16h a, v16h b, v8f c) {
  return __builtin_amdgcn_wmma_f32_16x16x32_f16(false, a, false, b, (short)0, c,
                                                false, false);
}

// A-matrix 16x32 f16 fragment: lane<16: row=lane, K=k0..k0+15 (contiguous);
// lane>=16: row=lane-16, K=k0+16..k0+31.  Two contiguous 16B chunks.
__device__ __forceinline__ v16h load_a_frag(const f16* tile, int ldk, int k0, int lane) {
  const f16* p = tile + (size_t)(lane & 15) * ldk + k0 + ((lane >> 4) << 4);
  union { v16h v; v8h h[2]; } u;
  u.h[0] = *(const v8h*)(p);
  u.h[1] = *(const v8h*)(p + 8);
  return u.v;
}

// B-matrix 32x16 f16 fragment: lane<16: col=n0+lane, K=k0..k0+15 (contiguous);
// lane>=16: col, K=k0+16..k0+31.  One 32B load from weights prepped [N][ldk].
__device__ __forceinline__ v16h load_b_frag(const f16* W, int ldk, int n0, int k0, int lane) {
  const f16* p = W + (size_t)(n0 + (lane & 15)) * ldk + k0 + ((lane >> 4) << 4);
  return *(const v16h*)(p);
}

// fast activations: single v_exp + v_rcp, no IEEE-div sequences on the scan path
__device__ __forceinline__ float sigf(float v) {
  return __builtin_amdgcn_rcpf(1.0f + __expf(-v));
}
__device__ __forceinline__ float tanhf_fast(float v) {
  return 2.0f * __builtin_amdgcn_rcpf(1.0f + __expf(-2.0f * v)) - 1.0f;
}

// ------------------------------------------------- encoder augmented weights
// gate tile g: 0=r (merged), 1=z (merged), 2=hn (h part + bhh), 3=in (x part + bih)
__device__ __forceinline__ float enc_w(int g, int hcol, int k,
                                       const float* Wih, const float* Whh,
                                       const float* bih, const float* bhh) {
  int grow = (g == 0) ? hcol : (g == 1) ? (ENCH + hcol) : (2 * ENCH + hcol);
  if (k < ENCH) {                       // h rows
    return (g == 3) ? 0.0f : Whh[(size_t)grow * ENCH + k];
  } else if (k < ENCH + 5) {            // x rows
    return (g == 2) ? 0.0f : Wih[(size_t)grow * 5 + (k - ENCH)];
  } else if (k == BIASK) {              // constant-1 row -> bias
    if (g <= 1) return bih[grow] + bhh[grow];
    if (g == 2) return bhh[grow];
    return bih[grow];
  }
  return 0.0f;                          // padding / zero slots 133..135
}

// ------------------------------------------------------- persistent encoder
__global__ __launch_bounds__(256) void enc_gru_persistent(
    const float* __restrict__ x, f16* __restrict__ Xh,
    const float* __restrict__ eWih, const float* __restrict__ eWhh,
    const float* __restrict__ ebih, const float* __restrict__ ebhh,
    f16* __restrict__ Actx) {
  __shared__ __align__(32) f16 H[MT * ELDK];  // 20 KB

  const int tid  = threadIdx.x;
  const int lane = tid & 31;
  const int w    = tid >> 5;          // wave 0..7, owns h-cols [16w,16w+16)
  const int b0   = blockIdx.x * MT;
  const int hcol = 16 * w + (lane & 15);

  // Build the 20 B fragments (4 gate tiles x 5 k-blocks) in registers, once.
  v16h Bf[4][5];
#pragma unroll
  for (int g = 0; g < 4; ++g)
#pragma unroll
    for (int kb = 0; kb < 5; ++kb) {
      v16h t;
#pragma unroll
      for (int e = 0; e < 16; ++e) {
        int k = kb * 32 + ((lane >> 4) << 4) + e;
        t[e] = (f16)enc_w(g, hcol, k, eWih, eWhh, ebih, ebhh);
      }
      Bf[g][kb] = t;
    }

  // Init augmented state: h=0, x(s=0), zero slots, bias col, pads.
  for (int idx = tid; idx < MT * ELDK; idx += 256) {
    int r = idx / ELDK, k = idx % ELDK;
    f16 v = (f16)0.0f;
    if (k == BIASK) v = (f16)1.0f;
    else if (k >= ENCH && k < ENCH + 5)
      v = (f16)x[(size_t)(b0 + r) * S_LEN * 5 + (k - ENCH)];
    H[idx] = v;
  }
  __syncthreads();

  v8f hprev[4];
#pragma unroll
  for (int rs = 0; rs < 4; ++rs) { v8f z = {}; hprev[rs] = z; }

#pragma unroll 1
  for (int s = 0; s < S_LEN; ++s) {
#pragma unroll
    for (int rs = 0; rs < 4; ++rs) {
      v8f acc[4];
#pragma unroll
      for (int g = 0; g < 4; ++g) { v8f z = {}; acc[g] = z; }
#pragma unroll
      for (int kb = 0; kb < 5; ++kb) {
        v16h a = load_a_frag(H + (size_t)rs * 16 * ELDK, ELDK, kb * 32, lane);
#pragma unroll
        for (int g = 0; g < 4; ++g) acc[g] = wmma16(a, Bf[g][kb], acc[g]);
      }
      // GRU gating directly on aligned fragments (biases already inside).
      v8f hn;
#pragma unroll
      for (int e = 0; e < 8; ++e) {
        float r = sigf(acc[0][e]);
        float z = sigf(acc[1][e]);
        float n = tanhf_fast(acc[3][e] + r * acc[2][e]);
        hn[e] = (1.0f - z) * n + z * hprev[rs][e];
      }
      hprev[rs] = hn;
    }
    __syncthreads();  // all H reads for step s complete

    // stage x(s+1): one async 16B global->LDS transfer per batch row
    if (s + 1 < S_LEN) {
#ifdef USE_ASYNC_LDS
      if (tid < MT) {
        f16* g = Xh + (((size_t)(b0 + tid)) * S_LEN + (s + 1)) * 8;
        __builtin_amdgcn_global_load_async_to_lds_b128(
            (AS1 v4i*)g, (AS3 v4i*)&H[(size_t)tid * ELDK + ENCH], 0, 0);
      }
#else
      for (int idx = tid; idx < MT * 8; idx += 256) {
        int r = idx >> 3, c = idx & 7;
        H[(size_t)r * ELDK + ENCH + c] =
            Xh[(((size_t)(b0 + r)) * S_LEN + (s + 1)) * 8 + c];
      }
#endif
      if (s + 2 < S_LEN && tid < MT)  // gfx1250 global_prefetch path
        __builtin_prefetch(&Xh[(((size_t)(b0 + tid)) * S_LEN + (s + 2)) * 8], 0, 1);
    }

    // write h' (this wave's 16 columns, all 64 rows) back to LDS
#pragma unroll
    for (int rs = 0; rs < 4; ++rs)
#pragma unroll
      for (int e = 0; e < 8; ++e) {
        int m = rs * 16 + e + ((lane >> 4) << 3);
        H[(size_t)m * ELDK + hcol] = (f16)hprev[rs][e];
      }
#ifdef USE_ASYNC_LDS
    __builtin_amdgcn_s_wait_asynccnt(0);
#endif
    __syncthreads();  // writes (incl. async x) visible for step s+1
  }

  // emit context rows into Actx (augmented: [ctx(128) | 1 | pad])
#pragma unroll
  for (int rs = 0; rs < 4; ++rs)
#pragma unroll
    for (int e = 0; e < 8; ++e) {
      int m = rs * 16 + e + ((lane >> 4) << 3);
      Actx[(size_t)(b0 + m) * CLDK + hcol] = (f16)hprev[rs][e];
    }
}

// --------------------------------------------------------- generic WMMA GEMM
// C[M x N](f32) = A[M x nkb*32](f16, lda) * B[N x nkb*32]^T (f16, ldb, K-contig)
// grid = (M/128, N/64); wave w handles row strip m0 = bx*128 + 16w.
__global__ __launch_bounds__(256) void gemm_wmma_f16(
    const f16* __restrict__ A, int lda, const f16* __restrict__ B, int ldb,
    float* __restrict__ C, int ldc, int nkb) {
  const int lane = threadIdx.x & 31;
  const int w    = threadIdx.x >> 5;
  const int m0   = blockIdx.x * 128 + w * 16;
  const int n0   = blockIdx.y * 64;

  v8f acc[4];
#pragma unroll
  for (int ct = 0; ct < 4; ++ct) { v8f z = {}; acc[ct] = z; }

  for (int kb = 0; kb < nkb; ++kb) {
    v16h a = load_a_frag(A + (size_t)m0 * lda, lda, kb * 32, lane);
#pragma unroll
    for (int ct = 0; ct < 4; ++ct) {
      v16h b = load_b_frag(B, ldb, n0 + ct * 16, kb * 32, lane);
      acc[ct] = wmma16(a, b, acc[ct]);
    }
  }
#pragma unroll
  for (int ct = 0; ct < 4; ++ct) {
    int n = n0 + ct * 16 + (lane & 15);
#pragma unroll
    for (int e = 0; e < 8; ++e) {
      int m = m0 + e + ((lane >> 4) << 3);
      C[(size_t)m * ldc + n] = acc[ct][e];
    }
  }
}

// ------------------------------------------------------------- small kernels
__global__ void fill_ones_f32(float* p, int n) {
  int i = blockIdx.x * blockDim.x + threadIdx.x;
  if (i < n) p[i] = 1.0f;
}

// pack x as f16 [B][S][8] (5 features + 3 zeros) for 16B async staging
__global__ void prep_x16(const float* __restrict__ x, f16* __restrict__ Xh) {
  int i = blockIdx.x * blockDim.x + threadIdx.x;
  if (i >= B_TOT * S_LEN * 8) return;
  int c = i & 7;
  int bs = i >> 3;
  Xh[i] = (c < 5) ? (f16)x[(size_t)bs * 5 + c] : (f16)0.0f;
}

__global__ void prep_dec_weights(const float* __restrict__ dWih,
                                 const float* __restrict__ dWhh,
                                 const float* __restrict__ dbih,
                                 const float* __restrict__ dbhh,
                                 f16* __restrict__ Wdec, f16* __restrict__ Wctx,
                                 float* __restrict__ wy) {
  int i = blockIdx.x * blockDim.x + threadIdx.x;
  if (i < 3 * DECH * DLDK) {  // [768][288] = [Whh | bhh | pad]
    int n = i / DLDK, k = i % DLDK;
    float v = 0.0f;
    if (k < DECH) v = dWhh[(size_t)n * DECH + k];
    else if (k == DECH) v = dbhh[n];
    Wdec[i] = (f16)v;
  }
  if (i < 3 * DECH * CLDK) {  // [768][160] = [Wih[:, :128] | bih | pad]
    int n = i / CLDK, k = i % CLDK;
    float v = 0.0f;
    if (k < ENCH) v = dWih[(size_t)n * (ENCH + 1) + k];
    else if (k == ENCH) v = dbih[n];
    Wctx[i] = (f16)v;
  }
  if (i < 3 * DECH) wy[i] = dWih[(size_t)i * (ENCH + 1) + ENCH];  // y column
}

__global__ void init_aug_state(f16* __restrict__ Haug, f16* __restrict__ Actx) {
  int i = blockIdx.x * blockDim.x + threadIdx.x;
  if (i < B_TOT * DLDK) {  // decoder h0 = ones; bias col = 1; pad = 0
    int k = i % DLDK;
    Haug[i] = (f16)((k <= DECH) ? 1.0f : 0.0f);
  }
  if (i < B_TOT * CLDK) {  // Actx aux cols (ctx cols written by encoder)
    int k = i % CLDK;
    if (k >= ENCH) Actx[i] = (f16)((k == ENCH) ? 1.0f : 0.0f);
  }
}

// per (b, j): GRU gating for decoder step s
__global__ void dec_gate(const float* __restrict__ G, const float* __restrict__ GI,
                         const float* __restrict__ wy, const float* __restrict__ y,
                         f16* __restrict__ Haug, int s) {
  int idx = blockIdx.x * blockDim.x + threadIdx.x;
  int b = idx >> 8, j = idx & 255;
  if (b >= B_TOT) return;
  float yt  = y[(size_t)b * S_LEN + s];
  size_t gb = (size_t)b * 3 * DECH;
  float gr  = G[gb + j]             + GI[gb + j]             + yt * wy[j];
  float gz  = G[gb + DECH + j]      + GI[gb + DECH + j]      + yt * wy[DECH + j];
  float hn  = G[gb + 2 * DECH + j];
  float gin = GI[gb + 2 * DECH + j] + yt * wy[2 * DECH + j];
  float r = sigf(gr), z = sigf(gz);
  float n = tanhf_fast(gin + r * hn);
  float h = (float)Haug[(size_t)b * DLDK + j];
  Haug[(size_t)b * DLDK + j] = (f16)((1.0f - z) * n + z * h);
}

// per (b, c): out = h' @ lin_W.T + lin_b
__global__ void dec_out(const f16* __restrict__ Haug, const float* __restrict__ linW,
                        const float* __restrict__ linb, float* __restrict__ out, int s) {
  int idx = blockIdx.x * blockDim.x + threadIdx.x;
  int b = idx >> 2, c = idx & 3;
  if (b >= B_TOT) return;
  float acc = linb[c];
  const f16* h = Haug + (size_t)b * DLDK;
  for (int k = 0; k < DECH; ++k) acc += (float)h[k] * linW[(size_t)c * DECH + k];
  out[((size_t)b * S_LEN + s) * 4 + c] = acc;
}

// ------------------------------------------------------------------- launch
extern "C" void kernel_launch(void* const* d_in, const int* in_sizes, int n_in,
                              void* d_out, int out_size, void* d_ws, size_t ws_size,
                              hipStream_t stream) {
  const float* x    = (const float*)d_in[0];
  const float* y    = (const float*)d_in[1];
  const float* eWih = (const float*)d_in[2];
  const float* eWhh = (const float*)d_in[3];
  const float* ebih = (const float*)d_in[4];
  const float* ebhh = (const float*)d_in[5];
  const float* dWih = (const float*)d_in[6];
  const float* dWhh = (const float*)d_in[7];
  const float* dbih = (const float*)d_in[8];
  const float* dbhh = (const float*)d_in[9];
  const float* linW = (const float*)d_in[10];
  const float* linb = (const float*)d_in[11];
  float* out = (float*)d_out;

  uint8_t* ws = (uint8_t*)d_ws;
  size_t off = 0;
  auto carve = [&](size_t bytes) -> void* {
    off = (off + 255) & ~(size_t)255;
    void* p = ws + off;
    off += bytes;
    return p;
  };
  f16*   Actx = (f16*)carve((size_t)B_TOT * CLDK * 2);
  f16*   Haug = (f16*)carve((size_t)B_TOT * DLDK * 2);
  f16*   Xh   = (f16*)carve((size_t)B_TOT * S_LEN * 8 * 2);
  f16*   Wdec = (f16*)carve((size_t)3 * DECH * DLDK * 2);
  f16*   Wctx = (f16*)carve((size_t)3 * DECH * CLDK * 2);
  float* wy   = (float*)carve((size_t)3 * DECH * 4);
  float* G    = (float*)carve((size_t)B_TOT * 3 * DECH * 4);
  float* GI   = (float*)carve((size_t)B_TOT * 3 * DECH * 4);

  {
    int n = B_TOT * S_LEN * 4;  // outputs default to 1.0
    fill_ones_f32<<<(n + 255) / 256, 256, 0, stream>>>(out, n);
  }
  {
    int n = B_TOT * S_LEN * 8;
    prep_x16<<<(n + 255) / 256, 256, 0, stream>>>(x, Xh);
  }
  {
    int n = 3 * DECH * DLDK;
    prep_dec_weights<<<(n + 255) / 256, 256, 0, stream>>>(dWih, dWhh, dbih, dbhh,
                                                          Wdec, Wctx, wy);
  }
  {
    int n = B_TOT * DLDK;
    init_aug_state<<<(n + 255) / 256, 256, 0, stream>>>(Haug, Actx);
  }

  // persistent fused encoder (writes context into Actx)
  enc_gru_persistent<<<B_TOT / MT, 256, 0, stream>>>(x, Xh, eWih, eWhh, ebih, ebhh,
                                                     Actx);

  // gi_const = [ctx|1] @ [Wih[:, :128]|bih]^T  (constant over decoder steps)
  gemm_wmma_f16<<<dim3(B_TOT / 128, (3 * DECH) / 64), 256, 0, stream>>>(
      Actx, CLDK, Wctx, CLDK, GI, 3 * DECH, CLDK / 32);

  for (int s = 0; s < 10; ++s) {
    // gh = [h|1] @ [Whh|bhh]^T
    gemm_wmma_f16<<<dim3(B_TOT / 128, (3 * DECH) / 64), 256, 0, stream>>>(
        Haug, DLDK, Wdec, DLDK, G, 3 * DECH, DLDK / 32);
    dec_gate<<<(B_TOT * DECH) / 256, 256, 0, stream>>>(G, GI, wy, y, Haug, s);
    dec_out<<<(B_TOT * 4 + 127) / 128, 128, 0, stream>>>(Haug, linW, linb, out, s);
  }
}
